// PrefillChunk1_71914932404735
// MI455X (gfx1250) — compile-verified, matrix operand order checked by reading, not run on hardware
//
// MI455X (gfx1250, CDNA5, wave32) transformer prefill.
// Strategy: bf16 WMMA (16x16x32) everywhere, fp32 accumulate.
//  - ~1.2GB fp32 weights streamed once, converted to bf16 tile-wise in LDS.
//  - GEMM tile 128x128, 8 waves x (32x64), double-buffered LDS.
//  - A-tile staged by the Tensor Data Mover (tensor_load_to_lds + s_wait_tensorcnt),
//    issued one tile ahead; B-tile staged+converted by VALU, overlapped with WMMA.
//  - Attention: flash-style per-wave 16 q-rows, online softmax, WMMA QK^T and PV.
//  - GELU uses the gfx1250 hardware tanh when available (branch-free fallback).

#include <hip/hip_runtime.h>

typedef __attribute__((ext_vector_type(16))) __bf16 v16bf;
typedef __attribute__((ext_vector_type(8)))  float  v8f;

#define NTOK   1024
#define DMODEL 2048
#define NH     8
#define NKV    2
#define HDIM   256
#define DFFC   4096
#define PLDC   256
#define NLC    30
#define NLAYER 8
#define EPSF   1e-6f

#if defined(__HIP_DEVICE_COMPILE__) && defined(__gfx1250__) && \
    __has_builtin(__builtin_amdgcn_tensor_load_to_lds) &&      \
    __has_builtin(__builtin_amdgcn_s_wait_tensorcnt)
#define HAVE_TDM 1
#else
#define HAVE_TDM 0
#endif

union FragBF { v16bf v; uint4 q[2]; };
union Pack2  { __bf16 b[2]; unsigned int u; };

__device__ __forceinline__ unsigned int pk2(float a, float b) {
  Pack2 t; t.b[0] = (__bf16)a; t.b[1] = (__bf16)b; return t.u;
}

// Branch-free tanh: hardware V_TANH_F32 if the builtin exists, else via v_exp_f32.
__device__ __forceinline__ float fast_tanh(float y) {
#if defined(__HIP_DEVICE_COMPILE__) && __has_builtin(__builtin_amdgcn_tanhf)
  return __builtin_amdgcn_tanhf(y);
#elif defined(__HIP_DEVICE_COMPILE__)
  float a = fabsf(y);
  float t = 1.f - 2.f / (__expf(2.f * a) + 1.f);   // tanh(|y|), exact identity
  return copysignf(t, y);
#else
  return tanhf(y);
#endif
}

__device__ __forceinline__ float gelu_tanh(float x) {
  float x3 = x * x * x;
  return 0.5f * x * (1.f + fast_tanh(0.7978845608028654f * (x + 0.044715f * x3)));
}

__device__ __forceinline__ float block_reduce_sum(float v, float* sred) {
  #pragma unroll
  for (int off = 16; off > 0; off >>= 1) v += __shfl_down(v, off, 32);
  int lane = threadIdx.x & 31, w = threadIdx.x >> 5;
  __syncthreads();
  if (lane == 0) sred[w] = v;
  __syncthreads();
  int nw = (blockDim.x + 31) >> 5;
  float total = 0.f;
  #pragma unroll
  for (int i = 0; i < 8; i++) if (i < nw) total += sred[i];
  return total;
}

// ---------------------------------------------------------------------------
// Generic GEMM: Y[M,E] = epi( X_bf16[M,K] * W_f32[E,K]^T * scale )
// Tile: 128(M) x 128(E), K-step 32. 8 waves (4x2), each 32x64 via 2x4 frags.
// ---------------------------------------------------------------------------
#define GTM 128
#define GTE 128
#define GTK 32
#define GLD (GTK + 8)   // row pad: 80B stride (64B data + 16B), 16B aligned

enum { EPI_NONE = 0, EPI_GELU = 1 };

#if HAVE_TDM
typedef unsigned int tdm_v4u __attribute__((ext_vector_type(4)));
typedef int tdm_v4i __attribute__((ext_vector_type(4)));
typedef int tdm_v8i __attribute__((ext_vector_type(8)));

// TDM: load a 128-row x 32-col bf16 tile (row stride K elems) into LDS with
// 16B padding after every 64B row (matches sA's GLD=40 layout).
// D# built per CDNA5 ISA ch.8: group0 {count=1, lds_addr, global_addr, type=2},
// group1 {data_size=2B, pad_enable, pad_interval=16DW, pad_amount=4DW,
//         tensor_dim0=K, tensor_dim1=M, tile=32x128, dim0_stride=K}.
__device__ __forceinline__ void tdm_load_tile(unsigned lds_off, const __bf16* g, int K) {
  unsigned long long ga = (unsigned long long)(uintptr_t)g;
  tdm_v4u g0;
  g0[0] = 1u;                                   // count=1, user desc
  g0[1] = lds_off;                              // lds_addr
  g0[2] = (unsigned)(ga & 0xffffffffu);         // global_addr[31:0]
  g0[3] = (unsigned)((ga >> 32) & 0x01ffffffu)  // global_addr[56:32]
          | (2u << 30);                         // type=2 (image)
  tdm_v8i g1;
  g1[0] = (int)((1u << 16) |                    // data_size = 2B
                (1u << 20) |                    // pad_enable
                (3u << 22) |                    // pad_interval: 16 DWORDs
                (3u << 25));                    // pad_amount: 4 DWORDs
  g1[1] = (int)((unsigned)(K & 0xffff) << 16);              // tensor_dim0[15:0]
  g1[2] = (int)(((unsigned)K >> 16) & 0xffffu) |            // tensor_dim0[31:16]
          (int)((unsigned)(NTOK & 0xffff) << 16);           // tensor_dim1[15:0]
  g1[3] = (int)(((unsigned)NTOK >> 16) & 0xffffu) |         // tensor_dim1[31:16]
          (int)(32u << 16);                                 // tile_dim0 = 32
  g1[4] = 128;                                              // tile_dim1 = 128
  g1[5] = K;                                                // dim0_stride[31:0]
  g1[6] = 0;
  g1[7] = 0;
  tdm_v4i z4 = {0, 0, 0, 0};
#if __clang_major__ >= 23
  tdm_v8i z8 = {0, 0, 0, 0, 0, 0, 0, 0};
  __builtin_amdgcn_tensor_load_to_lds(g0, g1, z4, z4, z8, 0);
#else
  __builtin_amdgcn_tensor_load_to_lds(g0, g1, z4, z4, 0);
#endif
}
#endif  // HAVE_TDM

__global__ __launch_bounds__(256, 1)
void k_gemm_bf16(const __bf16* __restrict__ X, const float* __restrict__ W,
                 float* __restrict__ Y, int M, int K, int E, int epi, float scale) {
  __shared__ __bf16 sA[2][GTM][GLD];
  __shared__ __bf16 sB[2][GTE][GLD];
  const int tid  = threadIdx.x;
  const int lane = tid & 31, wave = tid >> 5;
  const int hw   = lane >> 4, lrow = lane & 15;
  const int wm   = wave & 3,  we   = wave >> 2;     // 4x2 wave grid
  const int m0   = blockIdx.y * GTM;
  const int e0   = blockIdx.x * GTE;

  const int arow = tid >> 1, ach = (tid & 1) * 16;  // A fallback: 128 rows x 2 chunks
  const int brow = tid >> 1, bcs = (tid & 1) * 16;  // B: 128 rows x 2 chunks of 16 f32

  v8f c[2][4] = {};

  auto stage_b = [&](int k0, int buf) {
    const float4* gp = (const float4*)(W + (size_t)(e0 + brow) * K + k0 + bcs);
    float4 a0 = gp[0], a1 = gp[1], a2 = gp[2], a3 = gp[3];
    unsigned int* sp = (unsigned int*)&sB[buf][brow][bcs];
    sp[0] = pk2(a0.x, a0.y); sp[1] = pk2(a0.z, a0.w);
    sp[2] = pk2(a1.x, a1.y); sp[3] = pk2(a1.z, a1.w);
    sp[4] = pk2(a2.x, a2.y); sp[5] = pk2(a2.z, a2.w);
    sp[6] = pk2(a3.x, a3.y); sp[7] = pk2(a3.z, a3.w);
    __builtin_prefetch(W + (size_t)(e0 + brow) * K + k0 + GTK, 0, 0);
  };
#if !HAVE_TDM
  auto stage_a = [&](int k0, int buf) {
    const uint4* gp = (const uint4*)(X + (size_t)(m0 + arow) * K + k0 + ach);
    uint4 a0 = gp[0], a1 = gp[1];
    uint4* sp = (uint4*)&sA[buf][arow][ach];
    sp[0] = a0; sp[1] = a1;
  };
#endif

  // ---- prologue: tile 0 ----
  int buf = 0;
#if HAVE_TDM
  if (wave == 0) {
    tdm_load_tile((unsigned)(uintptr_t)&sA[0][0][0], X + (size_t)m0 * K, K);
  }
#else
  stage_a(0, 0);
#endif
  stage_b(0, 0);
#if HAVE_TDM
  if (wave == 0) __builtin_amdgcn_s_wait_tensorcnt(0);
#endif
  __syncthreads();

  for (int k0 = 0; k0 < K; k0 += GTK, buf ^= 1) {
    const int nk = k0 + GTK;
    if (nk < K) {
#if HAVE_TDM
      if (wave == 0)
        tdm_load_tile((unsigned)(uintptr_t)&sA[buf ^ 1][0][0],
                      X + (size_t)m0 * K + nk, K);
#else
      stage_a(nk, buf ^ 1);
#endif
      stage_b(nk, buf ^ 1);
    }
    // ---- compute from sA/sB[buf] ----
    FragBF af[2], bf[4];
    #pragma unroll
    for (int i = 0; i < 2; i++) {
      const __bf16* p = &sA[buf][wm * 32 + i * 16 + lrow][0];
      af[i].q[0] = *(const uint4*)(p + hw * 8);
      af[i].q[1] = *(const uint4*)(p + 16 + hw * 8);
    }
    #pragma unroll
    for (int j = 0; j < 4; j++) {
      const __bf16* p = &sB[buf][we * 64 + j * 16 + lrow][0];
      bf[j].q[0] = *(const uint4*)(p + hw * 8);
      bf[j].q[1] = *(const uint4*)(p + 16 + hw * 8);
    }
    #pragma unroll
    for (int i = 0; i < 2; i++)
      #pragma unroll
      for (int j = 0; j < 4; j++)
        c[i][j] = __builtin_amdgcn_wmma_f32_16x16x32_bf16(
            false, af[i].v, false, bf[j].v, (short)0, c[i][j], false, false);
#if HAVE_TDM
    if (wave == 0 && nk < K) __builtin_amdgcn_s_wait_tensorcnt(0);
#endif
    __syncthreads();
  }

  // epilogue + store. C layout: row = reg + 8*hw, col = lrow (per 16x16 frag)
  #pragma unroll
  for (int i = 0; i < 2; i++)
    #pragma unroll
    for (int j = 0; j < 4; j++)
      #pragma unroll
      for (int r = 0; r < 8; r++) {
        float v = c[i][j][r] * scale;
        if (epi == EPI_GELU) v = gelu_tanh(v);
        int grow = m0 + wm * 32 + i * 16 + hw * 8 + r;
        int gcol = e0 + we * 64 + j * 16 + lrow;
        Y[(size_t)grow * E + gcol] = v;
      }
}

// ---------------------------------------------------------------------------
// Flash attention (no sqrt scale; analytic causal mask = -1e9, matches ref).
// Q [NH,N,HD] bf16, K [NKV,N,HD] bf16, VT [NKV,HD,N] bf16 -> O [N, NH*HD] bf16
// ---------------------------------------------------------------------------
#define AT_WAVES 4
__global__ __launch_bounds__(128, 1)
void k_attn(const __bf16* __restrict__ Q, const __bf16* __restrict__ Kb,
            const __bf16* __restrict__ VT, __bf16* __restrict__ O, int N) {
  __shared__ __bf16 sQ[64][HDIM + 8];
  __shared__ __bf16 sP[AT_WAVES][16][GTK + 8];
  const int h  = blockIdx.x;
  const int q0 = blockIdx.y * 64;
  const int kv = h >> 2;                       // GQA repeat 4
  const int tid = threadIdx.x, lane = tid & 31, wave = tid >> 5;
  const int hw = lane >> 4, lrow = lane & 15;

  for (int idx = tid; idx < 64 * (HDIM / 8); idx += 128) {
    int r = idx >> 5, cc = (idx & 31) * 8;
    *(uint4*)&sQ[r][cc] = *(const uint4*)(Q + ((size_t)h * N + q0 + r) * HDIM + cc);
  }
  __syncthreads();

  const int qr = wave * 16;
  v8f acc[16] = {};
  float mrow[8], lsum[8];
  #pragma unroll
  for (int r = 0; r < 8; r++) { mrow[r] = -3e38f; lsum[r] = 0.f; }

  for (int kt = 0; kt < N; kt += 32) {
    v8f s[2] = {};
    #pragma unroll
    for (int kb = 0; kb < HDIM / 32; kb++) {
      FragBF aq;
      const __bf16* qp = &sQ[qr + lrow][kb * 32];
      aq.q[0] = *(const uint4*)(qp + hw * 8);
      aq.q[1] = *(const uint4*)(qp + 16 + hw * 8);
      #pragma unroll
      for (int j = 0; j < 2; j++) {
        FragBF bk;
        const __bf16* kp = Kb + ((size_t)kv * N + kt + j * 16 + lrow) * HDIM + kb * 32;
        bk.q[0] = *(const uint4*)(kp + hw * 8);
        bk.q[1] = *(const uint4*)(kp + 16 + hw * 8);
        s[j] = __builtin_amdgcn_wmma_f32_16x16x32_bf16(
            false, aq.v, false, bk.v, (short)0, s[j], false, false);
      }
    }
    float corr[8];
    #pragma unroll
    for (int r = 0; r < 8; r++) {
      int qrow = q0 + qr + hw * 8 + r;
      float mx = mrow[r];
      #pragma unroll
      for (int j = 0; j < 2; j++) {
        int key = kt + j * 16 + lrow;
        if (key > qrow) s[j][r] += -1e9f;
        mx = fmaxf(mx, s[j][r]);
      }
      #pragma unroll
      for (int off = 1; off < 16; off <<= 1) mx = fmaxf(mx, __shfl_xor(mx, off, 32));
      corr[r] = __expf(mrow[r] - mx);
      mrow[r] = mx;
      float psum = 0.f;
      #pragma unroll
      for (int j = 0; j < 2; j++) {
        float p = __expf(s[j][r] - mx);
        s[j][r] = p; psum += p;
      }
      #pragma unroll
      for (int off = 1; off < 16; off <<= 1) psum += __shfl_xor(psum, off, 32);
      lsum[r] = lsum[r] * corr[r] + psum;
    }
    #pragma unroll
    for (int jd = 0; jd < 16; jd++)
      #pragma unroll
      for (int r = 0; r < 8; r++) acc[jd][r] *= corr[r];

    #pragma unroll
    for (int j = 0; j < 2; j++)
      #pragma unroll
      for (int r = 0; r < 8; r++)
        sP[wave][hw * 8 + r][j * 16 + lrow] = (__bf16)s[j][r];
    FragBF ap;
    {
      const __bf16* pp = &sP[wave][lrow][0];
      ap.q[0] = *(const uint4*)(pp + hw * 8);
      ap.q[1] = *(const uint4*)(pp + 16 + hw * 8);
    }
    #pragma unroll
    for (int jd = 0; jd < 16; jd++) {
      FragBF bv;
      const __bf16* vp = VT + ((size_t)kv * HDIM + jd * 16 + lrow) * N + kt;
      bv.q[0] = *(const uint4*)(vp + hw * 8);
      bv.q[1] = *(const uint4*)(vp + 16 + hw * 8);
      acc[jd] = __builtin_amdgcn_wmma_f32_16x16x32_bf16(
          false, ap.v, false, bv.v, (short)0, acc[jd], false, false);
    }
  }

  #pragma unroll
  for (int jd = 0; jd < 16; jd++)
    #pragma unroll
    for (int r = 0; r < 8; r++) {
      int n = q0 + qr + hw * 8 + r;
      O[(size_t)n * (NH * HDIM) + h * HDIM + jd * 16 + lrow] =
          (__bf16)(acc[jd][r] / lsum[r]);
    }
}

// ---------------------------------------------------------------------------
// Elementwise / norm kernels
// ---------------------------------------------------------------------------
__global__ __launch_bounds__(HDIM)
void k_norm_rope(const float* __restrict__ in, const float* __restrict__ w,
                 const float* __restrict__ cosb, const float* __restrict__ sinb,
                 __bf16* __restrict__ outb, float* __restrict__ outf, int N, int Hx) {
  __shared__ float sred[8];
  __shared__ float sy[HDIM];
  int n = blockIdx.x, h = blockIdx.y, d = threadIdx.x;
  float v = in[((size_t)n * Hx + h) * HDIM + d];
  float ss = block_reduce_sum(v * v, sred);
  float y = v * rsqrtf(ss / HDIM + EPSF) * (1.f + w[d]);
  sy[d] = y;
  __syncthreads();
  float rot = (d < HDIM / 2) ? -sy[d + HDIM / 2] : sy[d - HDIM / 2];
  float o = y * cosb[(size_t)n * HDIM + d] + rot * sinb[(size_t)n * HDIM + d];
  size_t oi = ((size_t)h * N + n) * HDIM + d;
  outb[oi] = (__bf16)o;
  if (outf) outf[oi] = o;
}

__global__ __launch_bounds__(HDIM)
void k_vnorm(const float* __restrict__ in, __bf16* __restrict__ vtb,
             float* __restrict__ outf, int N) {
  __shared__ float sred[8];
  int n = blockIdx.x, h = blockIdx.y, d = threadIdx.x;
  float v = in[((size_t)n * NKV + h) * HDIM + d];
  float ss = block_reduce_sum(v * v, sred);
  float y = v * rsqrtf(ss / HDIM + EPSF);
  outf[((size_t)h * N + n) * HDIM + d] = y;
  vtb[((size_t)h * HDIM + d) * N + n] = (__bf16)y;
}

__global__ __launch_bounds__(256)
void k_rmsnorm_bf16(const float* __restrict__ in, const float* __restrict__ w,
                    __bf16* __restrict__ outb, int D) {
  __shared__ float sred[8];
  int n = blockIdx.x, t = threadIdx.x;
  const float* row = in + (size_t)n * D;
  float vv[8], ss = 0.f;
  #pragma unroll
  for (int i = 0; i < 8; i++) { vv[i] = row[t + i * 256]; ss += vv[i] * vv[i]; }
  ss = block_reduce_sum(ss, sred);
  float inv = rsqrtf(ss / D + EPSF);
  #pragma unroll
  for (int i = 0; i < 8; i++) {
    int idx = t + i * 256;
    outb[(size_t)n * D + idx] = (__bf16)(vv[i] * inv * (1.f + w[idx]));
  }
}

__global__ __launch_bounds__(256)
void k_res_rmsnorm(const float* __restrict__ res, const float* __restrict__ g,
                   const float* __restrict__ w, float* __restrict__ hout,
                   const float* __restrict__ scal, int D) {
  __shared__ float sred[8];
  int n = blockIdx.x, t = threadIdx.x;
  const float* gr = g + (size_t)n * D;
  float vv[8], ss = 0.f;
  #pragma unroll
  for (int i = 0; i < 8; i++) { vv[i] = gr[t + i * 256]; ss += vv[i] * vv[i]; }
  ss = block_reduce_sum(ss, sred);
  float inv = rsqrtf(ss / D + EPSF);
  float s = scal ? *scal : 1.f;
  #pragma unroll
  for (int i = 0; i < 8; i++) {
    int idx = t + i * 256;
    hout[(size_t)n * D + idx] =
        (res[(size_t)n * D + idx] + vv[i] * inv * (1.f + w[idx])) * s;
  }
}

__global__ __launch_bounds__(PLDC)
void k_ple_norm(const float* __restrict__ proj, const float* __restrict__ wn,
                const float* __restrict__ raw, float* __restrict__ ple) {
  __shared__ float sred[8];
  int n = blockIdx.x, l = blockIdx.y, p = threadIdx.x;
  size_t idx = ((size_t)n * NLC + l) * PLDC + p;
  float v = proj[idx];
  float ss = block_reduce_sum(v * v, sred);
  float y = v * rsqrtf(ss / PLDC + EPSF) * (1.f + wn[p]);
  ple[idx] = (y + raw[idx]) * 0.7071067811865476f;
}

__global__ void k_mul_bf16(const float* __restrict__ a, const float* __restrict__ b,
                           __bf16* __restrict__ o, long total, int C, int bstride, int boff) {
  long i = (long)blockIdx.x * blockDim.x + threadIdx.x;
  if (i >= total) return;
  long n = i / C; int c = (int)(i - n * C);
  o[i] = (__bf16)(a[i] * b[n * (long)bstride + boff + c]);
}

__global__ void k_cvt_bf16(const float* __restrict__ in, __bf16* __restrict__ out, long n4) {
  long i = (long)blockIdx.x * blockDim.x + threadIdx.x;
  if (i >= n4) return;
  float4 v = ((const float4*)in)[i];
  ((uint2*)out)[i] = make_uint2(pk2(v.x, v.y), pk2(v.z, v.w));
}

// ---------------------------------------------------------------------------
// Host orchestration
// ---------------------------------------------------------------------------
static inline void gemm(const __bf16* X, const float* W, float* Y, int M, int K,
                        int E, int epi, float scale, hipStream_t s) {
  dim3 g(E / GTE, M / GTM);
  k_gemm_bf16<<<g, 256, 0, s>>>(X, W, Y, M, K, E, epi, scale);
}

extern "C" void kernel_launch(void* const* d_in, const int* in_sizes, int n_in,
                              void* d_out, int out_size, void* d_ws, size_t ws_size,
                              hipStream_t stream) {
  const int N = NTOK, D = DMODEL;
  const float* hs       = (const float*)d_in[0];
  // d_in[1] causal_mask unused: mask computed analytically (identical values)
  const float* raw      = (const float*)d_in[2];
  const float* cos_s    = (const float*)d_in[3];
  const float* sin_s    = (const float*)d_in[4];
  const float* cos_f    = (const float*)d_in[5];
  const float* sin_f    = (const float*)d_in[6];
  const float* ple_mw   = (const float*)d_in[7];
  const float* ple_nw   = (const float*)d_in[8];
  const float* ln_in    = (const float*)d_in[9];
  const float* q_w      = (const float*)d_in[10];
  const float* k_w      = (const float*)d_in[11];
  const float* v_w      = (const float*)d_in[12];
  const float* o_w      = (const float*)d_in[13];
  const float* q_norm   = (const float*)d_in[14];
  const float* k_norm   = (const float*)d_in[15];
  const float* ln_pattn = (const float*)d_in[16];
  const float* ln_preff = (const float*)d_in[17];
  const float* gate_w   = (const float*)d_in[18];
  const float* up_w     = (const float*)d_in[19];
  const float* down_w   = (const float*)d_in[20];
  const float* ln_pff   = (const float*)d_in[21];
  const float* plegw    = (const float*)d_in[22];
  const float* plepw    = (const float*)d_in[23];
  const float* ln_pple  = (const float*)d_in[24];
  const float* lscal    = (const float*)d_in[25];

  float* out_h   = (float*)d_out;
  float* out_ple = out_h + (size_t)N * D;
  float* out_Ks  = out_ple + (size_t)N * NLC * PLDC;
  float* out_Vs  = out_Ks + (size_t)NLAYER * NKV * N * HDIM;

  char* wp = (char*)d_ws;
  auto alloc = [&](size_t bytes) {
    void* p = (void*)wp; wp += (bytes + 255) & ~(size_t)255; return p;
  };
  float*  hbuf = (float*)alloc((size_t)N * D * 4);
  __bf16* xbf  = (__bf16*)alloc((size_t)N * D * 2);
  __bf16* mbf  = (__bf16*)alloc((size_t)N * DFFC * 2);
  __bf16* aobf = (__bf16*)alloc((size_t)N * NH * HDIM * 2);
  __bf16* qbf  = (__bf16*)alloc((size_t)NH * N * HDIM * 2);
  __bf16* kbf  = (__bf16*)alloc((size_t)NKV * N * HDIM * 2);
  __bf16* vtbf = (__bf16*)alloc((size_t)NKV * HDIM * N * 2);
  float*  t0   = (float*)alloc((size_t)N * DFFC * 4);
  float*  t1   = (float*)alloc((size_t)N * DFFC * 4);
  float*  t2   = (float*)alloc((size_t)N * D * 4);
  float*  tple = t0;  // 30MB aliases t0+t1 (32MB); only used pre-loop

  const long nd4 = (long)N * D / 4;
  const int CV = 256;

  // ---- PLE precompute ----
  k_cvt_bf16<<<dim3((nd4 + CV - 1) / CV), CV, 0, stream>>>(hs, xbf, nd4);
  gemm(xbf, ple_mw, tple, N, D, NLC * PLDC, EPI_NONE, 0.022097086912079608f, stream);
  k_ple_norm<<<dim3(N, NLC), PLDC, 0, stream>>>(tple, ple_nw, raw, out_ple);

  hipMemcpyAsync(hbuf, hs, (size_t)N * D * 4, hipMemcpyDeviceToDevice, stream);

  for (int li = 0; li < NLAYER; li++) {
    const float* cosb = (li == 7) ? cos_f : cos_s;
    const float* sinb = (li == 7) ? sin_f : sin_s;
    // ---- attention ----
    k_rmsnorm_bf16<<<N, 256, 0, stream>>>(hbuf, ln_in + (size_t)li * D, xbf, D);
    gemm(xbf, q_w + (size_t)li * NH * HDIM * D, t0, N, D, NH * HDIM, EPI_NONE, 1.f, stream);
    gemm(xbf, k_w + (size_t)li * NKV * HDIM * D, t1, N, D, NKV * HDIM, EPI_NONE, 1.f, stream);
    gemm(xbf, v_w + (size_t)li * NKV * HDIM * D, t2, N, D, NKV * HDIM, EPI_NONE, 1.f, stream);
    k_norm_rope<<<dim3(N, NH), HDIM, 0, stream>>>(
        t0, q_norm + (size_t)li * HDIM, cosb, sinb, qbf, nullptr, N, NH);
    k_norm_rope<<<dim3(N, NKV), HDIM, 0, stream>>>(
        t1, k_norm + (size_t)li * HDIM, cosb, sinb, kbf,
        out_Ks + (size_t)li * NKV * N * HDIM, N, NKV);
    k_vnorm<<<dim3(N, NKV), HDIM, 0, stream>>>(
        t2, vtbf, out_Vs + (size_t)li * NKV * N * HDIM, N);
    k_attn<<<dim3(NH, N / 64), 128, 0, stream>>>(qbf, kbf, vtbf, aobf, N);
    gemm(aobf, o_w + (size_t)li * D * NH * HDIM, t2, N, NH * HDIM, D, EPI_NONE, 1.f, stream);
    k_res_rmsnorm<<<N, 256, 0, stream>>>(hbuf, t2, ln_pattn + (size_t)li * D, hbuf, nullptr, D);
    // ---- FFN ----
    k_rmsnorm_bf16<<<N, 256, 0, stream>>>(hbuf, ln_preff + (size_t)li * D, xbf, D);
    gemm(xbf, gate_w + (size_t)li * DFFC * D, t0, N, D, DFFC, EPI_GELU, 1.f, stream);
    gemm(xbf, up_w + (size_t)li * DFFC * D, t1, N, D, DFFC, EPI_NONE, 1.f, stream);
    {
      long tot = (long)N * DFFC;
      k_mul_bf16<<<dim3((tot + CV - 1) / CV), CV, 0, stream>>>(t0, t1, mbf, tot, DFFC, DFFC, 0);
    }
    gemm(mbf, down_w + (size_t)li * D * DFFC, t2, N, DFFC, D, EPI_NONE, 1.f, stream);
    k_res_rmsnorm<<<N, 256, 0, stream>>>(hbuf, t2, ln_pff + (size_t)li * D, hbuf, nullptr, D);
    // ---- PLE path (raw h input) ----
    k_cvt_bf16<<<dim3((nd4 + CV - 1) / CV), CV, 0, stream>>>(hbuf, xbf, nd4);
    gemm(xbf, plegw + (size_t)li * PLDC * D, t0, N, D, PLDC, EPI_GELU, 1.f, stream);
    {
      long tot = (long)N * PLDC;
      k_mul_bf16<<<dim3((tot + CV - 1) / CV), CV, 0, stream>>>(
          t0, out_ple, mbf, tot, PLDC, NLC * PLDC, li * PLDC);
    }
    gemm(mbf, plepw + (size_t)li * D * PLDC, t2, N, PLDC, D, EPI_NONE, 1.f, stream);
    k_res_rmsnorm<<<N, 256, 0, stream>>>(hbuf, t2, ln_pple + (size_t)li * D, hbuf, lscal + li, D);
  }

  hipMemcpyAsync(out_h, hbuf, (size_t)N * D * 4, hipMemcpyDeviceToDevice, stream);
}
